// PositionAttentionModule_54090818126711
// MI455X (gfx1250) — matrix operand order repaired and checked by reading, not compiled, not run on hardware
//
#include <hip/hip_runtime.h>
#include <hip/hip_bf16.h>

// ---------------------------------------------------------------------------
// Types for CDNA5 WMMA (wave32): 16x16x32 bf16 -> f32
// ---------------------------------------------------------------------------
typedef __attribute__((ext_vector_type(16))) __bf16 v16bf;
typedef __attribute__((ext_vector_type(8)))  float  v8f;
typedef __attribute__((ext_vector_type(4)))  unsigned int v4ui;
typedef __attribute__((ext_vector_type(8)))  int v8i;
typedef __attribute__((ext_vector_type(4)))  int v4i;

union FragU { uint4 u[2]; v16bf v; };
union PackU { __bf16 h[8]; uint4 u; };

__device__ __forceinline__ float bf2f(__bf16 x) {
    union { __bf16 b; unsigned short s; } i{x};
    union { unsigned u; float f; } o{(unsigned)i.s << 16};
    return o.f;
}

__device__ __forceinline__ __bf16 f2bf(float f) {
    union { float f; unsigned u; } a{f};
    unsigned r = a.u + 0x7FFFu + ((a.u >> 16) & 1u);   // round-to-nearest-even
    union { unsigned short s; __bf16 b; } o{(unsigned short)(r >> 16)};
    return o.b;
}

// Load a 16-element bf16 fragment as two 16B chunks, `gap` elements apart.
// A-operand (16x32, row-major [M,K]): base = row*K + k + half*8, gap = 16
// B-operand (32x16, act stored [Ncol,K]): base = col*K + k + half*8, gap = 16
__device__ __forceinline__ v16bf load_frag(const __bf16* p, int gap) {
    FragU f;
    f.u[0] = *reinterpret_cast<const uint4*>(p);
    f.u[1] = *reinterpret_cast<const uint4*>(p + gap);
    return f.v;
}

// ---------------------------------------------------------------------------
// Tensor Data Mover: stage one K-tile (16 rows x 512 bf16) into LDS.
// 2-D descriptor, data_size=2B, pad 4 DWORDs every 256 DWORDs (1 KB row)
// so the LDS row stride is 520 bf16 = 1040 B (bank-conflict-free b128 reads).
// 6-arg builtin form (clang-23 / therock-10.0 headers).
// ---------------------------------------------------------------------------
__device__ __forceinline__ void tdm_load_k_tile(const __bf16* gsrc, unsigned lds_off) {
    unsigned long long ga = (unsigned long long)(size_t)gsrc;
    v4ui g0;
    g0[0] = 1u;                                        // count=1, user mode
    g0[1] = lds_off;                                   // lds_addr
    g0[2] = (unsigned)(ga & 0xffffffffu);              // global_addr[31:0]
    g0[3] = (unsigned)((ga >> 32) & 0x01ffffffu) | (2u << 30);  // ga[56:32] | type=2
    v8i g1;
    g1[0] = (int)((1u << 16)      // data_size = 1 -> 2 bytes
                | (1u << 20)      // pad_enable
                | (7u << 22)      // pad_interval code 7 -> 256 DWORDs
                | (3u << 25));    // pad_amount code 3 -> 4 DWORDs (16 B)
    g1[1] = (int)(512u << 16);    // tensor_dim0[15:0] = 512 (in [31:16])
    g1[2] = (int)(4096u << 16);   // tensor_dim0 hi = 0 | tensor_dim1 lo16 = 4096
    g1[3] = (int)(512u << 16);    // tensor_dim1 hi = 0 | tile_dim0 = 512
    g1[4] = 16;                   // tile_dim1 = 16 | tile_dim2 = 0
    g1[5] = 512;                  // tensor_dim0_stride[31:0] = 512
    g1[6] = 0;
    g1[7] = 0;
    v4i z4 = {0, 0, 0, 0};
    v8i z8 = {0, 0, 0, 0, 0, 0, 0, 0};
    __builtin_amdgcn_tensor_load_to_lds(g0, g1, z4, z4, z8, 0);
}

// ---------------------------------------------------------------------------
// BN fold: scale = g * rsqrt(v+eps), bias = b - m*scale
// ---------------------------------------------------------------------------
__global__ void bn_prep_kernel(const float* __restrict__ g, const float* __restrict__ b,
                               const float* __restrict__ m, const float* __restrict__ v,
                               float* __restrict__ scale, float* __restrict__ bias, int n) {
    int i = blockIdx.x * blockDim.x + threadIdx.x;
    if (i < n) {
        float s = g[i] * rsqrtf(v[i] + 1e-5f);
        scale[i] = s;
        bias[i]  = b[i] - m[i] * s;
    }
}

// f32 -> bf16 elementwise (weights)
__global__ void cvt_f2bf_kernel(const float* __restrict__ in, __bf16* __restrict__ out, long n) {
    long i = (long)blockIdx.x * blockDim.x + threadIdx.x;
    if (i < n) out[i] = f2bf(in[i]);
}

// feature [B,C,N] f32 -> [B,N,C] bf16 via LDS tile transpose
__global__ void transpose_cvt_kernel(const float* __restrict__ in, __bf16* __restrict__ out,
                                     int C, int Nsp) {
    __shared__ float tile[32][33];
    int b  = blockIdx.z;
    int n0 = blockIdx.x * 32;
    int c0 = blockIdx.y * 32;
    const float* ib = in + (size_t)b * C * Nsp;
    __bf16* ob = out + (size_t)b * C * Nsp;
    int tx = threadIdx.x, ty = threadIdx.y;
    #pragma unroll
    for (int i = 0; i < 32; i += 8)
        tile[ty + i][tx] = ib[(size_t)(c0 + ty + i) * Nsp + n0 + tx];
    __syncthreads();
    #pragma unroll
    for (int i = 0; i < 32; i += 8)
        ob[(size_t)(n0 + ty + i) * C + c0 + tx] = f2bf(tile[tx][ty + i]);
}

// ---------------------------------------------------------------------------
// Generic batched WMMA GEMM:  out[M,Nn] = A[M,K] * Bt[Nn,K]^T  (per batch)
// Wave computes 16(M) x 64(Nn). K unrolled by 64 with two structurally
// alternating fragment sets (no rotation movs); loads for the next half-step
// are issued before the wmma group of the current one, so VMEM overlaps the
// matrix pipe with partial loadcnt waits. K must be a multiple of 64.
// ---------------------------------------------------------------------------
__global__ void __launch_bounds__(256)
gemm_wmma_bf16(const __bf16* __restrict__ A, long a_bs,
               const __bf16* __restrict__ Bt, long b_bs,
               const float* __restrict__ scale, long s_bs,
               const float* __restrict__ bias,
               const __bf16* __restrict__ resid, long r_bs,
               __bf16* __restrict__ outNM, long onm_bs,   // [Nn, M] bf16 (packed b128 stores)
               __bf16* __restrict__ outMN, long omn_bs,   // [M, Nn] bf16
               float*  __restrict__ outMNf, long omf_bs,  // [M, Nn] f32
               int M, int Nn, int K, int nb, int do_relu) {
    int gw   = (int)((blockIdx.x * blockDim.x + threadIdx.x) >> 5);
    int lane = threadIdx.x & 31;
    int hl   = lane >> 4;        // half: 0 or 1
    int l15  = lane & 15;
    int n64  = Nn >> 6;
    int m16  = M >> 4;
    int perb = n64 * m16;
    if (gw >= perb * nb) return;
    int b   = gw / perb;
    int rem = gw - b * perb;
    int mt  = rem / n64;
    int nt  = rem - mt * n64;

    const __bf16* Ab = A  + (size_t)b * a_bs;
    const __bf16* Bb = Bt + (size_t)b * b_bs;

    v8f acc[4] = {};
    const __bf16* arowp = Ab + (size_t)(mt * 16 + l15) * K + hl * 8;
    const __bf16* bcolp = Bb + (size_t)(nt * 64 + l15) * K + hl * 8;

    // prologue: fragment set 0 at k = 0
    v16bf af0 = load_frag(arowp, 16);
    v16bf bf0[4];
    #pragma unroll
    for (int j = 0; j < 4; j++) bf0[j] = load_frag(bcolp + (size_t)(j * 16) * K, 16);

    for (int k = 0; k < K; k += 64) {
        // load set 1 at k+32 (always in range: K multiple of 64)
        int k1 = k + 32;
        v16bf af1 = load_frag(arowp + k1, 16);
        v16bf bf1[4];
        #pragma unroll
        for (int j = 0; j < 4; j++) bf1[j] = load_frag(bcolp + (size_t)(j * 16) * K + k1, 16);

        // consume set 0
        #pragma unroll
        for (int j = 0; j < 4; j++)
            acc[j] = __builtin_amdgcn_wmma_f32_16x16x32_bf16(
                false, af0, false, bf0[j], (short)0, acc[j], false, false);

        // load set 0 at k+64 (wrap to 0 on the last trip; result unused)
        int k2 = (k + 64 < K) ? (k + 64) : 0;
        af0 = load_frag(arowp + k2, 16);
        #pragma unroll
        for (int j = 0; j < 4; j++) bf0[j] = load_frag(bcolp + (size_t)(j * 16) * K + k2, 16);

        // consume set 1
        #pragma unroll
        for (int j = 0; j < 4; j++)
            acc[j] = __builtin_amdgcn_wmma_f32_16x16x32_bf16(
                false, af1, false, bf1[j], (short)0, acc[j], false, false);
    }

    #pragma unroll
    for (int j = 0; j < 4; j++) {
        int ncol = nt * 64 + j * 16 + l15;
        PackU pk;
        #pragma unroll
        for (int r = 0; r < 8; r++) {
            int mrow = mt * 16 + hl * 8 + r;
            float vv = acc[j][r];
            if (scale)  vv *= scale[(size_t)b * s_bs + mrow];
            if (bias)   vv += bias[mrow];
            if (resid)  vv += bf2f(resid[(size_t)b * r_bs + (size_t)mrow * Nn + ncol]);
            if (do_relu) vv = fmaxf(vv, 0.0f);
            pk.h[r] = f2bf(vv);
            if (outMN)  outMN [(size_t)b * omn_bs + (size_t)mrow * Nn + ncol] = pk.h[r];
            if (outMNf) outMNf[(size_t)b * omf_bs + (size_t)mrow * Nn + ncol] = vv;
        }
        if (outNM)
            *reinterpret_cast<uint4*>(outNM + (size_t)b * onm_bs +
                                      (size_t)ncol * M + mt * 16 + hl * 8) = pk.u;
    }
}

// ---------------------------------------------------------------------------
// Attention scores + softmax (unnormalized). One workgroup = 8 waves, each
// owning 16 query rows of the SAME batch; all waves sweep the same K-tiles,
// which are staged into LDS by the Tensor Data Mover (double-buffered,
// TENSORcnt-synchronized). Pass 1: row max. Pass 2: P' = exp(S-max) + sums.
// Row stride in LDS = 520 bf16 (TDM pad) -> conflict-free ds_load_b128.
// ---------------------------------------------------------------------------
__global__ void __launch_bounds__(256)
attn_softmax_wmma(const __bf16* __restrict__ Q, const __bf16* __restrict__ Kc,
                  __bf16* __restrict__ P, float* __restrict__ inv_l,
                  int Nq, int Rc, int nb) {
    __shared__ __align__(32) __bf16 shK[2][16 * 520];

    int tid  = threadIdx.x;
    int wid  = tid >> 5;
    int lane = tid & 31;
    int hl   = lane >> 4;
    int l15  = lane & 15;
    int qtiles = Nq >> 4;
    int gw = blockIdx.x * 8 + wid;
    int b  = gw / qtiles;
    int qt = gw - b * qtiles;

    const __bf16* Qb = Q  + (size_t)b * Nq * Rc;
    const __bf16* Kb = Kc + (size_t)b * Nq * Rc;
    __bf16* Pb = P + (size_t)b * Nq * Nq;

    unsigned lds0 = (unsigned)(size_t)&shK[0][0];
    unsigned lds1 = (unsigned)(size_t)&shK[1][0];

    // Preload all 16 Q fragments (K-dim = Rc = 512 -> 16 k-steps of 32).
    v16bf qf[16];
    const __bf16* qp = Qb + (size_t)(qt * 16 + l15) * Rc + hl * 8;
    #pragma unroll
    for (int s = 0; s < 16; s++) qf[s] = load_frag(qp + s * 32, 16);

    float rmax[8];
    #pragma unroll
    for (int r = 0; r < 8; r++) rmax[r] = -3.0e38f;
    float rsum[8] = {0.f, 0.f, 0.f, 0.f, 0.f, 0.f, 0.f, 0.f};
    const float LOG2E = 1.4426950408889634f;

    for (int pass = 0; pass < 2; pass++) {
        int buf = 0;
        if (wid == 0) tdm_load_k_tile(Kb, lds0);           // prologue: tile 0
        for (int mt = 0; mt < qtiles; mt++) {
            if (mt + 1 < qtiles) {
                if (wid == 0)
                    tdm_load_k_tile(Kb + (size_t)(mt + 1) * 16 * Rc, buf ? lds0 : lds1);
                __builtin_amdgcn_s_wait_tensorcnt(1);      // current tile done
            } else {
                __builtin_amdgcn_s_wait_tensorcnt(0);
            }
            __syncthreads();                               // publish LDS tile

            v8f acc = {};
            const __bf16* kl = &shK[buf][l15 * 520 + hl * 8];
            #pragma unroll
            for (int s = 0; s < 16; s++) {
                v16bf kf = load_frag(kl + s * 32, 16);
                acc = __builtin_amdgcn_wmma_f32_16x16x32_bf16(
                    false, qf[s], false, kf, (short)0, acc, false, false);
            }
            __syncthreads();                               // done reading buf

            if (pass == 0) {
                #pragma unroll
                for (int r = 0; r < 8; r++) rmax[r] = fmaxf(rmax[r], acc[r]);
            } else {
                #pragma unroll
                for (int r = 0; r < 8; r++) {
                    float e = exp2f((acc[r] - rmax[r]) * LOG2E);
                    rsum[r] += e;
                    Pb[(size_t)(qt * 16 + hl * 8 + r) * Nq + mt * 16 + l15] = f2bf(e);
                }
            }
            buf ^= 1;
        }
        if (pass == 0) {
            #pragma unroll
            for (int msk = 1; msk < 16; msk <<= 1) {
                #pragma unroll
                for (int r = 0; r < 8; r++)
                    rmax[r] = fmaxf(rmax[r], __shfl_xor(rmax[r], msk));
            }
        }
    }

    #pragma unroll
    for (int msk = 1; msk < 16; msk <<= 1) {
        #pragma unroll
        for (int r = 0; r < 8; r++)
            rsum[r] += __shfl_xor(rsum[r], msk);
    }
    if (l15 == 0) {
        #pragma unroll
        for (int r = 0; r < 8; r++)
            inv_l[(size_t)b * Nq + qt * 16 + hl * 8 + r] = 1.0f / rsum[r];
    }
}

// ---------------------------------------------------------------------------
// Host orchestration
// ---------------------------------------------------------------------------
extern "C" void kernel_launch(void* const* d_in, const int* in_sizes, int n_in,
                              void* d_out, int out_size, void* d_ws, size_t ws_size,
                              hipStream_t stream) {
    const int B = 4, C = 2048, R = 512, Nsp = 4096;

    const float* feature = (const float*)d_in[0];
    const float* Wf[5];  const float* gP[5]; const float* bP[5];
    const float* mP[5];  const float* vP[5];
    for (int l = 0; l < 5; l++) {
        Wf[l] = (const float*)d_in[1 + l * 5 + 0];
        gP[l] = (const float*)d_in[1 + l * 5 + 1];
        bP[l] = (const float*)d_in[1 + l * 5 + 2];
        mP[l] = (const float*)d_in[1 + l * 5 + 3];
        vP[l] = (const float*)d_in[1 + l * 5 + 4];
    }
    const int cin[5]  = {C, R, R, R, R};
    const int cout[5] = {R, R, R, R, C};

    char* ws = (char*)d_ws;
    size_t off = 0;
    auto alloc = [&](size_t bytes) -> char* {
        char* p = ws + off;
        off += (bytes + 255) & ~(size_t)255;
        return p;
    };

    __bf16* featbf = (__bf16*)alloc((size_t)B * Nsp * C * 2);      // [B][n][C]
    __bf16* Wbf[5];
    float*  scl[5]; float* bia[5];
    for (int l = 0; l < 5; l++) {
        Wbf[l] = (__bf16*)alloc((size_t)cout[l] * cin[l] * 2);     // [M,K]
        scl[l] = (float*)alloc((size_t)cout[l] * 4);
        bia[l] = (float*)alloc((size_t)cout[l] * 4);
    }
    __bf16* f_nm = (__bf16*)alloc((size_t)B * Nsp * R * 2);        // [B][n][R]
    __bf16* q_nm = (__bf16*)alloc((size_t)B * Nsp * R * 2);
    __bf16* k_nm = (__bf16*)alloc((size_t)B * Nsp * R * 2);
    __bf16* v_cm = (__bf16*)alloc((size_t)B * R * Nsp * 2);        // [B][c][m] channel-first
    __bf16* Pws  = (__bf16*)alloc((size_t)B * Nsp * Nsp * 2);      // [B][q][m] unnormalized exp
    float*  invl = (float*)alloc((size_t)B * Nsp * 4);
    __bf16* fsum = (__bf16*)alloc((size_t)B * Nsp * R * 2);        // [B][n][R] f + upd

    // 1) BN fold + weight conversion
    for (int l = 0; l < 5; l++) {
        bn_prep_kernel<<<(cout[l] + 255) / 256, 256, 0, stream>>>(
            gP[l], bP[l], mP[l], vP[l], scl[l], bia[l], cout[l]);
        long ne = (long)cout[l] * cin[l];
        cvt_f2bf_kernel<<<(int)((ne + 255) / 256), 256, 0, stream>>>(Wf[l], Wbf[l], ne);
    }

    // 2) feature [B,C,N] f32 -> [B,N,C] bf16
    transpose_cvt_kernel<<<dim3(Nsp / 32, C / 32, B), dim3(32, 8), 0, stream>>>(
        feature, featbf, C, Nsp);

    auto gemm = [&](const __bf16* A, long a_bs, const __bf16* Bt, long b_bs,
                    const float* scale, long s_bs, const float* bias,
                    const __bf16* resid, long r_bs,
                    __bf16* oNM, long onm_bs, __bf16* oMN, long omn_bs,
                    float* oMNf, long omf_bs,
                    int M, int Nn, int K, int relu) {
        int waves  = B * (M / 16) * (Nn / 64);
        int blocks = (waves * 32 + 255) / 256;
        gemm_wmma_bf16<<<blocks, 256, 0, stream>>>(
            A, a_bs, Bt, b_bs, scale, s_bs, bias, resid, r_bs,
            oNM, onm_bs, oMN, omn_bs, oMNf, omf_bs, M, Nn, K, B, relu);
    };

    // 3) r-CBR: f = relu(bn(Wr @ feat))   -> [n, R]
    gemm(Wbf[0], 0, featbf, (long)Nsp * C, scl[0], 0, bia[0], nullptr, 0,
         f_nm, (long)Nsp * R, nullptr, 0, nullptr, 0, R, Nsp, C, 1);

    // 4) q, k (channels-last), v (channel-first for U GEMM B-operand)
    gemm(Wbf[1], 0, f_nm, (long)Nsp * R, scl[1], 0, bia[1], nullptr, 0,
         q_nm, (long)Nsp * R, nullptr, 0, nullptr, 0, R, Nsp, R, 1);
    gemm(Wbf[2], 0, f_nm, (long)Nsp * R, scl[2], 0, bia[2], nullptr, 0,
         k_nm, (long)Nsp * R, nullptr, 0, nullptr, 0, R, Nsp, R, 1);
    gemm(Wbf[3], 0, f_nm, (long)Nsp * R, scl[3], 0, bia[3], nullptr, 0,
         nullptr, 0, v_cm, (long)R * Nsp, nullptr, 0, R, Nsp, R, 1);

    // 5) attention scores + softmax (TDM-staged K tiles; exact grid: 1024 waves)
    {
        int waves  = B * (Nsp / 16);
        int blocks = waves / 8;
        attn_softmax_wmma<<<blocks, 256, 0, stream>>>(q_nm, k_nm, Pws, invl, Nsp, R, B);
    }

    // 6) fsum[q,c] = f[q,c] + inv_l[q] * (P' @ V)   (normalization folded in)
    gemm(Pws, (long)Nsp * Nsp, v_cm, (long)R * Nsp, invl, (long)Nsp, nullptr,
         f_nm, (long)Nsp * R, nullptr, 0, fsum, (long)Nsp * R, nullptr, 0,
         Nsp, R, Nsp, 0);

    // 7) u-CBR: out = relu(bn(Wu @ fsum)) -> f32 [B, C, N] directly to d_out
    gemm(Wbf[4], 0, fsum, (long)Nsp * R, scl[4], 0, bia[4], nullptr, 0,
         nullptr, 0, nullptr, 0, (float*)d_out, (long)C * Nsp, C, Nsp, R, 1);

    (void)in_sizes; (void)n_in; (void)out_size; (void)ws_size;
}
